// RNN_71803263254601
// MI455X (gfx1250) — matrix operand order repaired and checked by reading, not compiled, not run on hardware
//
#include <hip/hip_runtime.h>
#include <math.h>

// Problem sizes (fixed by the reference)
#define Bn 64
#define Tn 512
#define Dn 512
#define Hn 512
#define BT (Bn * Tn)
#define NB_REC 8      // persistent blocks for the recurrent kernel (co-resident)
#define BS_LDW 520    // LDS row stride for B panel: 512 elems + 16B pad -> conflict-free

typedef __bf16 bf16;
typedef __attribute__((ext_vector_type(16))) __bf16 v16bf;
typedef __attribute__((ext_vector_type(8)))  __bf16 v8bf;
typedef __attribute__((ext_vector_type(8)))  float  v8f;
typedef __attribute__((ext_vector_type(4)))  unsigned v4u;
typedef __attribute__((ext_vector_type(8)))  int      v8i;
typedef __attribute__((ext_vector_type(4)))  int      v4i;

// ---------------- WMMA fragment loaders (contiguous k = 16h + s, same for A and B) --------

// A-matrix 16x32 bf16 (MxK), row-major, leading dim lda.
// lanes 0-15: row M=lane, K = k0+[0..15]
// lanes 16-31: row M=lane-16, K = k0+[16..31]
static __device__ __forceinline__ v16bf load_a_frag(const bf16* __restrict__ A,
                                                    int lda, int m0, int k0, int lane) {
  int r  = m0 + (lane & 15);
  int kb = k0 + ((lane >> 4) << 4);
  const bf16* p = A + (size_t)r * lda + kb;
  v8bf lo = *(const v8bf*)(p);
  v8bf hi = *(const v8bf*)(p + 8);
  return __builtin_shufflevector(lo, hi, 0,1,2,3,4,5,6,7,8,9,10,11,12,13,14,15);
}

// B-matrix 32x16 bf16 (KxN), B = W^T with W (N x K) row-major, ldw = K.
// lanes 0-15: col N=lane, K = k0+[0..15]; lanes 16-31: col N=lane-16, K = k0+[16..31]
static __device__ __forceinline__ v16bf load_b_frag(const bf16* __restrict__ W,
                                                    int ldw, int n0, int k0, int lane) {
  int n  = n0 + (lane & 15);
  int kb = k0 + ((lane >> 4) << 4);
  return *(const v16bf*)(W + (size_t)n * ldw + kb);
}

// ---------------- Device-scope grid barrier (monotonic counter) --------------

static __device__ __forceinline__ void grid_barrier(unsigned* ctr, unsigned target) {
  __syncthreads();
  if (threadIdx.x == 0) {
    __threadfence();
    __hip_atomic_fetch_add(ctr, 1u, __ATOMIC_ACQ_REL, __HIP_MEMORY_SCOPE_AGENT);
    while (__hip_atomic_load(ctr, __ATOMIC_ACQUIRE, __HIP_MEMORY_SCOPE_AGENT) < target) {
      __builtin_amdgcn_s_sleep(2);
    }
  }
  __syncthreads();
  __threadfence();
}

// ---------------- Kernels ----------------------------------------------------

__global__ void k_f32_to_bf16(const float* __restrict__ src, bf16* __restrict__ dst, int n) {
  int i = blockIdx.x * blockDim.x + threadIdx.x;
  if (i < n) dst[i] = (bf16)src[i];
}

__global__ void k_zero_ctr(unsigned* ctr) {
  if (threadIdx.x < 8) ctr[threadIdx.x] = 0u;
}

// C[M x 512] = A[M x 512](bf16) @ W[512 x 512](bf16, row-major, used transposed)
//            + bias1[n] + bias2[n]   (fp32 out)
// Block: 256 threads = 8 waves; tile 128 rows x 64 cols.
// The 64x512 W panel for this block is DMA'd into LDS once by the Tensor Data
// Mover (padded 16B per 1024B row -> bank-conflict-free B-fragment reads),
// then every wave reads B fragments via ds_load while A streams from global.
__global__ void k_xproj_gemm(const bf16* __restrict__ A, const bf16* __restrict__ W,
                             const float* __restrict__ b1, const float* __restrict__ b2,
                             float* __restrict__ C) {
  __shared__ __align__(128) bf16 Bs[64 * BS_LDW];   // ~66.5 KB

  const int lane = threadIdx.x & 31;
  const int wave = threadIdx.x >> 5;
  const int m0 = blockIdx.x * 128 + wave * 16;
  const int n0 = blockIdx.y * 64;

  // --- TDM: load W[n0 .. n0+64) x [0..512) into LDS (wave 0 only; scalar branch
  // via readfirstlane because TDM ignores EXEC).
  if (__builtin_amdgcn_readfirstlane(threadIdx.x) < 32) {
    unsigned lds_off = (unsigned)(size_t)(void*)&Bs[0];
    unsigned long long gaddr = (unsigned long long)(const void*)(W + (size_t)n0 * Dn);
    // D# group 0: count=1 | lds_addr | global_addr[56:0] | type=2
    v4u g0 = { 1u,
               lds_off,
               (unsigned)(gaddr & 0xFFFFFFFFu),
               (unsigned)((gaddr >> 32) & 0x01FFFFFFu) | (2u << 30) };
    // D# group 1: data_size=2B, pad_enable, pad_interval=256 DWORDs (code 7),
    // pad_amount=4 DWORDs (code 3); tensor_dim0=512, tensor_dim1=512,
    // tile_dim0=512, tile_dim1=64, tensor_dim0_stride=512.
    v8i g1 = { (int)((1u << 16) | (1u << 20) | (7u << 22) | (3u << 25)),
               (int)(((unsigned)Dn & 0xFFFFu) << 16),   // tensor_dim0 lo
               (int)(((unsigned)Hn & 0xFFFFu) << 16),   // dim0 hi=0 | tensor_dim1 lo
               (int)(((unsigned)Dn & 0xFFFFu) << 16),   // dim1 hi=0 | tile_dim0=512
               64,                                      // tile_dim1=64 | tile_dim2=0
               Dn,                                      // tensor_dim0_stride lo
               0, 0 };
    v4i g2 = { 0, 0, 0, 0 };
    v4i g3 = { 0, 0, 0, 0 };
    v8i g4 = { 0, 0, 0, 0, 0, 0, 0, 0 };
    __builtin_amdgcn_tensor_load_to_lds(g0, g1, g2, g3, g4, 0);
    __builtin_amdgcn_s_wait_tensorcnt(0);
  }
  __syncthreads();

  const int nl  = lane & 15;
  const int khi = (lane >> 4) << 4;

  v8f acc0 = {}, acc1 = {}, acc2 = {}, acc3 = {};
  for (int k0 = 0; k0 < Dn; k0 += 32) {
    v16bf a = load_a_frag(A, Dn, m0, k0, lane);
    #pragma unroll
    for (int j = 0; j < 4; ++j) {
      const bf16* p = &Bs[(j * 16 + nl) * BS_LDW + k0 + khi];
      v8bf blo = *(const v8bf*)(p);
      v8bf bhi = *(const v8bf*)(p + 8);
      v16bf bb = __builtin_shufflevector(blo, bhi, 0,1,2,3,4,5,6,7,8,9,10,11,12,13,14,15);
      if (j == 0) acc0 = __builtin_amdgcn_wmma_f32_16x16x32_bf16(false, a, false, bb, (short)0, acc0, false, false);
      if (j == 1) acc1 = __builtin_amdgcn_wmma_f32_16x16x32_bf16(false, a, false, bb, (short)0, acc1, false, false);
      if (j == 2) acc2 = __builtin_amdgcn_wmma_f32_16x16x32_bf16(false, a, false, bb, (short)0, acc2, false, false);
      if (j == 3) acc3 = __builtin_amdgcn_wmma_f32_16x16x32_bf16(false, a, false, bb, (short)0, acc3, false, false);
    }
  }

  const int hi8 = (lane >> 4) * 8;
  #pragma unroll
  for (int i = 0; i < 8; ++i) {
    const size_t r = (size_t)(m0 + hi8 + i);
    int c;
    c = n0 +  0 + nl; C[r * Hn + c] = acc0[i] + b1[c] + b2[c];
    c = n0 + 16 + nl; C[r * Hn + c] = acc1[i] + b1[c] + b2[c];
    c = n0 + 32 + nl; C[r * Hn + c] = acc2[i] + b1[c] + b2[c];
    c = n0 + 48 + nl; C[r * Hn + c] = acc3[i] + b1[c] + b2[c];
  }
}

// Persistent recurrent layer: h_t = tanh(xproj[:,t,:] + h_{t-1} @ Whh^T)
// NB_REC blocks x 512 threads (16 waves); W_hh fragments register-resident for
// all 512 steps; xproj loads hoisted ahead of the WMMA chain; next step's
// xproj prefetched (global_prefetch_b8) during current compute.
__global__ void k_rnn_rec(const bf16* __restrict__ Whh,    // H x H bf16
                          const float* __restrict__ xproj, // B x T x H fp32
                          bf16* __restrict__ seq,          // B x T x H bf16 (out)
                          bf16* __restrict__ hbuf,         // 2 x B x H bf16 (ping-pong)
                          float* __restrict__ hfin,        // B x H fp32 (final h)
                          unsigned* __restrict__ ctr) {
  const int lane = threadIdx.x & 31;
  const int wave = threadIdx.x >> 5;
  const int m0   = (wave >> 2) * 16;
  const int nG   = blockIdx.x * 64 + (wave & 3) * 16;
  const int nl   = lane & 15;
  const int hi8  = (lane >> 4) * 8;

  // Hoist W_hh fragments for this wave's 16 output columns (time-invariant).
  v16bf bw[16];
  #pragma unroll
  for (int kk = 0; kk < 16; ++kk) bw[kk] = load_b_frag(Whh, Hn, nG, kk * 32, lane);

  // h_{-1} = 0 in buffer 0 (each block zeroes its own 64 columns).
  for (int idx = threadIdx.x; idx < Bn * 64; idx += blockDim.x) {
    int r = idx >> 6;
    int c = blockIdx.x * 64 + (idx & 63);
    hbuf[(size_t)r * Hn + c] = (bf16)0.0f;
  }
  grid_barrier(ctr, (unsigned)NB_REC);

  for (int t = 0; t < Tn; ++t) {
    const bf16* hread  = hbuf + (size_t)(t & 1) * (Bn * Hn);
    bf16*       hwrite = hbuf + (size_t)((t + 1) & 1) * (Bn * Hn);

    // Warm next step's xproj rows into cache while this step computes.
    if (t + 1 < Tn) {
      const float* pf = xproj + (size_t)(m0 + nl) * (Tn * Hn) + (size_t)(t + 1) * Hn + nG;
      __builtin_prefetch(pf, 0, 0);
    }

    // Hoist this step's xproj loads ahead of the WMMA chain.
    float xp[8];
    #pragma unroll
    for (int i = 0; i < 8; ++i) {
      const size_t off = (size_t)(m0 + hi8 + i) * (Tn * Hn) + (size_t)t * Hn + (nG + nl);
      xp[i] = xproj[off];
    }

    v8f acc = {};
    #pragma unroll
    for (int kk = 0; kk < 16; ++kk) {
      v16bf a = load_a_frag(hread, Hn, m0, kk * 32, lane);
      acc = __builtin_amdgcn_wmma_f32_16x16x32_bf16(false, a, false, bw[kk], (short)0, acc, false, false);
    }

    #pragma unroll
    for (int i = 0; i < 8; ++i) {
      const int r = m0 + hi8 + i;
      const int c = nG + nl;
      const size_t off = (size_t)r * (Tn * Hn) + (size_t)t * Hn + c;
      float v = tanhf(acc[i] + xp[i]);
      bf16 vb = (bf16)v;
      seq[off] = vb;
      hwrite[(size_t)r * Hn + c] = vb;
      if (t == Tn - 1) hfin[(size_t)r * Hn + c] = v;
    }
    grid_barrier(ctr, (unsigned)(NB_REC * (t + 2)));
  }
}

// out[b] = dot(h_final[b,:], W_fc[0,:]) + b_fc[0]
__global__ void k_fc(const float* __restrict__ hfin, const float* __restrict__ Wfc,
                     const float* __restrict__ bfc, float* __restrict__ out) {
  int b = threadIdx.x;
  if (b < Bn) {
    float s = bfc[0];
    for (int h = 0; h < Hn; ++h) s += hfin[b * Hn + h] * Wfc[h];
    out[b] = s;
  }
}

// ---------------- Launch -----------------------------------------------------

extern "C" void kernel_launch(void* const* d_in, const int* in_sizes, int n_in,
                              void* d_out, int out_size, void* d_ws, size_t ws_size,
                              hipStream_t stream) {
  const float* x    = (const float*)d_in[0];
  const float* Wih0 = (const float*)d_in[1];
  const float* Whh0 = (const float*)d_in[2];
  const float* bih0 = (const float*)d_in[3];
  const float* bhh0 = (const float*)d_in[4];
  const float* Wih1 = (const float*)d_in[5];
  const float* Whh1 = (const float*)d_in[6];
  const float* bih1 = (const float*)d_in[7];
  const float* bhh1 = (const float*)d_in[8];
  const float* Wfc  = (const float*)d_in[9];
  const float* bfc  = (const float*)d_in[10];
  float* out = (float*)d_out;

  // Workspace layout (all offsets 256B-aligned)
  char* ws = (char*)d_ws;
  const size_t OFF_XPROJ = 0;                                   // B*T*H*4  = 64 MB
  const size_t OFF_XBF   = OFF_XPROJ + (size_t)BT * Hn * 4;     // B*T*D*2  = 32 MB
  const size_t OFF_SEQ   = OFF_XBF   + (size_t)BT * Dn * 2;     // B*T*H*2  = 32 MB
  const size_t OFF_HBF   = OFF_SEQ   + (size_t)BT * Hn * 2;     // 2*B*H*2
  const size_t OFF_HFP   = OFF_HBF   + (size_t)2 * Bn * Hn * 2; // B*H*4
  const size_t OFF_WBF   = OFF_HFP   + (size_t)Bn * Hn * 4;     // 4 weight mats bf16
  const size_t OFF_CTR   = OFF_WBF   + (size_t)4 * Hn * Dn * 2;

  float*    xproj = (float*)   (ws + OFF_XPROJ);
  bf16*     xbf   = (bf16*)    (ws + OFF_XBF);
  bf16*     seqb  = (bf16*)    (ws + OFF_SEQ);
  bf16*     hbuf  = (bf16*)    (ws + OFF_HBF);
  float*    hfin  = (float*)   (ws + OFF_HFP);
  bf16*     wih0b = (bf16*)    (ws + OFF_WBF + 0 * (size_t)Hn * Dn * 2);
  bf16*     whh0b = (bf16*)    (ws + OFF_WBF + 1 * (size_t)Hn * Dn * 2);
  bf16*     wih1b = (bf16*)    (ws + OFF_WBF + 2 * (size_t)Hn * Dn * 2);
  bf16*     whh1b = (bf16*)    (ws + OFF_WBF + 3 * (size_t)Hn * Dn * 2);
  unsigned* ctr   = (unsigned*)(ws + OFF_CTR);

  // bf16 conversions
  k_f32_to_bf16<<<(BT * Dn + 255) / 256, 256, 0, stream>>>(x, xbf, BT * Dn);
  k_f32_to_bf16<<<(Hn * Dn + 255) / 256, 256, 0, stream>>>(Wih0, wih0b, Hn * Dn);
  k_f32_to_bf16<<<(Hn * Hn + 255) / 256, 256, 0, stream>>>(Whh0, whh0b, Hn * Hn);
  k_f32_to_bf16<<<(Hn * Hn + 255) / 256, 256, 0, stream>>>(Wih1, wih1b, Hn * Hn);
  k_f32_to_bf16<<<(Hn * Hn + 255) / 256, 256, 0, stream>>>(Whh1, whh1b, Hn * Hn);

  const dim3 gemm_grid(BT / 128, Hn / 64);

  // Layer 0
  k_zero_ctr<<<1, 32, 0, stream>>>(ctr);
  k_xproj_gemm<<<gemm_grid, 256, 0, stream>>>(xbf, wih0b, bih0, bhh0, xproj);
  k_rnn_rec<<<NB_REC, 512, 0, stream>>>(whh0b, xproj, seqb, hbuf, hfin, ctr);

  // Layer 1
  k_zero_ctr<<<1, 32, 0, stream>>>(ctr);
  k_xproj_gemm<<<gemm_grid, 256, 0, stream>>>(seqb, wih1b, bih1, bhh1, xproj);
  k_rnn_rec<<<NB_REC, 512, 0, stream>>>(whh1b, xproj, seqb, hbuf, hfin, ctr);

  // Final FC
  k_fc<<<1, 64, 0, stream>>>(hfin, Wfc, bfc, out);
}
